// Attention_566935683261
// MI455X (gfx1250) — matrix operand order repaired and checked by reading, not compiled, not run on hardware
//
#include <hip/hip_runtime.h>
#include <hip/hip_bf16.h>
#include <math.h>

// ---------------------------------------------------------------------------
// Attention layer for MI455X (gfx1250): bf16 WMMA everywhere.
//   B=2, S=2048, DIM=1024, H=16, HD=64
// Tiling chosen to fit the 128-VGPR/wave budget exactly (no scratch spills):
//   GEMM: 64x32 per wave;  attention: 16 queries per wave.
// ---------------------------------------------------------------------------

#define Bb   2
#define Ss   2048
#define DIM  1024
#define NH   16
#define HD   64
#define BS   (Bb*Ss)          // 4096 rows
#define QKVN (3*DIM)          // 3072

typedef __attribute__((ext_vector_type(16))) __bf16 v16bf;
typedef __attribute__((ext_vector_type(8)))  float  v8f;

struct U32x8 { uint4 a, b; };

// Round-to-nearest-even fp32 -> bf16 (as raw ushort)
static __device__ inline unsigned short f2bf(float f) {
    unsigned int u = __builtin_bit_cast(unsigned int, f);
    u += 0x7FFFu + ((u >> 16) & 1u);
    return (unsigned short)(u >> 16);
}

// Build a 16-half WMMA fragment from two 16-byte chunks.
static __device__ inline v16bf load_frag(const unsigned short* p0,
                                         const unsigned short* p1) {
    U32x8 u;
    u.a = *(const uint4*)p0;
    u.b = *(const uint4*)p1;
    return __builtin_bit_cast(v16bf, u);
}

static __device__ inline v8f wmma_bf16(v16bf a, v16bf b, v8f c) {
    return __builtin_amdgcn_wmma_f32_16x16x32_bf16(
        /*neg_a=*/false, a, /*neg_b=*/false, b,
        /*c_mod=*/(short)0, c, /*reuse_a=*/false, /*reuse_b=*/false);
}

// ---------------------------------------------------------------------------
// Kernel 1: fp32 -> bf16 elementwise
// ---------------------------------------------------------------------------
__global__ void k_to_bf16(const float* __restrict__ in,
                          unsigned short* __restrict__ out, int n) {
    int i = blockIdx.x * blockDim.x + threadIdx.x;
    if (i < n) out[i] = f2bf(in[i]);
}

// ---------------------------------------------------------------------------
// Kernel 2/5: C[M,N] = A[M,K] * W[N,K]^T   (bf16 in, f32 out)
// Per-wave tile: 64(M) x 32(N).  Block = 4 waves -> 64 x 128.
// grid = (N/128, M/64).  8 accumulators + frags == 128 VGPRs, no spill.
// ---------------------------------------------------------------------------
__global__ void k_gemm_bf16(const unsigned short* __restrict__ A,
                            const unsigned short* __restrict__ W,
                            float* __restrict__ C, int M, int N, int K) {
    const int wave = threadIdx.x >> 5;
    const int lane = threadIdx.x & 31;
    const int lm   = lane & 15;
    const int hi   = lane >> 4;
    const int m0   = blockIdx.y * 64;
    const int n0   = (blockIdx.x * 4 + wave) * 32;

    v8f acc[4][2] = {};
    const unsigned short* arow[4];
#pragma unroll
    for (int mi = 0; mi < 4; ++mi)
        arow[mi] = A + (size_t)(m0 + mi*16 + lm) * K;
    const unsigned short* brow[2];
#pragma unroll
    for (int ni = 0; ni < 2; ++ni)
        brow[ni] = W + (size_t)(n0 + ni*16 + lm) * K;

    for (int k = 0; k < K; k += 32) {
        v16bf bf[2];
#pragma unroll
        for (int ni = 0; ni < 2; ++ni)
            bf[ni] = load_frag(brow[ni] + k + hi*8,
                               brow[ni] + k + 16 + hi*8);
#pragma unroll
        for (int mi = 0; mi < 4; ++mi) {
            v16bf af = load_frag(arow[mi] + k + hi*8,
                                 arow[mi] + k + 16 + hi*8);
#pragma unroll
            for (int ni = 0; ni < 2; ++ni)
                acc[mi][ni] = wmma_bf16(af, bf[ni], acc[mi][ni]);
        }
    }
#pragma unroll
    for (int mi = 0; mi < 4; ++mi)
#pragma unroll
        for (int ni = 0; ni < 2; ++ni) {
            float* cp = C + (size_t)(m0 + mi*16 + hi*8) * N + n0 + ni*16 + lm;
#pragma unroll
            for (int v = 0; v < 8; ++v) cp[(size_t)v * N] = acc[mi][ni][v];
        }
}

// ---------------------------------------------------------------------------
// Kernel 3: RoPE on q,k + relayout to bf16.
//  Q,K : [B,H,S,64] bf16    V: transposed [B,H,64,S] bf16
//  tid -> ((b*S+s)*H + h)*32 + j  (j = rotation pair index)
// ---------------------------------------------------------------------------
__global__ void k_rope(const float* __restrict__ qkv,
                       const float* __restrict__ freqs,
                       unsigned short* __restrict__ Qo,
                       unsigned short* __restrict__ Ko,
                       unsigned short* __restrict__ Vt) {
    int tid = blockIdx.x * blockDim.x + threadIdx.x;
    if (tid >= BS * NH * 32) return;
    const int j  = tid & 31;
    const int h  = (tid >> 5) & 15;
    const int bs = tid >> 9;          // b*S + s
    const int s  = bs & (Ss - 1);
    const int b  = bs >> 11;

    const float* base = qkv + (size_t)bs * QKVN;
    const float  c  = freqs[(s*32 + j)*2 + 0];
    const float  sn = freqs[(s*32 + j)*2 + 1];
    const int    col = h*HD + 2*j;

    float qr = base[col],         qi = base[col + 1];
    float kr = base[DIM + col],   ki = base[DIM + col + 1];
    float vr = base[2*DIM + col], vi = base[2*DIM + col + 1];

    float qor = qr*c - qi*sn, qoi = qi*c + qr*sn;
    float kor = kr*c - ki*sn, koi = ki*c + kr*sn;

    const size_t bh   = (size_t)(b*NH + h);
    const size_t qoff = (bh*Ss + s)*HD + 2*j;
    Qo[qoff]   = f2bf(qor); Qo[qoff+1] = f2bf(qoi);
    Ko[qoff]   = f2bf(kor); Ko[qoff+1] = f2bf(koi);
    const size_t voff = (bh*HD + 2*j)*Ss + s;
    Vt[voff]      = f2bf(vr);
    Vt[voff + Ss] = f2bf(vi);
}

// ---------------------------------------------------------------------------
// Kernel 4: flash attention. One wave per (b,h, 16-query tile).
// Key loop in steps of 32. Scores via WMMA (K-dim = HD = 64, 2 steps),
// online softmax, P transposed C->A layout via per-wave LDS tile,
// y accumulated via WMMA (4 x 16-wide d chunks).
// Y written as bf16 rows [B*S, H*64] ready for the output GEMM.
// grid = 512 blocks x 256 threads (8 waves, 4096 waves total)
// ---------------------------------------------------------------------------
__global__ void k_attn(const unsigned short* __restrict__ Q,
                       const unsigned short* __restrict__ Kk,
                       const unsigned short* __restrict__ Vt,
                       unsigned short* __restrict__ Y) {
    __shared__ __align__(16) unsigned short plds[8][16 * 32];

    const int wave = threadIdx.x >> 5;
    const int lane = threadIdx.x & 31;
    const int lm   = lane & 15;
    const int hi   = lane >> 4;
    const int gw   = blockIdx.x * 8 + wave;   // 0..4095
    const int qt   = gw & 127;                // 128 q-tiles per (b,h)
    const int bh   = gw >> 7;                 // 0..31
    const int q0   = qt * 16;
    const int b    = bh >> 4;
    const int h    = bh & 15;

    const unsigned short* qbase = Q  + (size_t)bh * Ss * HD;
    const unsigned short* kbase = Kk + (size_t)bh * Ss * HD;
    const unsigned short* vbase = Vt + (size_t)bh * HD * Ss;

    // Q fragments for this tile: K-dim = d (64) -> two K=32 fragments.
    const unsigned short* qrow = qbase + (size_t)(q0 + lm) * HD;
    v16bf qf0 = load_frag(qrow +      hi*8, qrow + 16 + hi*8);
    v16bf qf1 = load_frag(qrow + 32 + hi*8, qrow + 48 + hi*8);

    float mrun[8], lrun[8];
    v8f   yacc[4] = {};
#pragma unroll
    for (int v = 0; v < 8; ++v) { mrun[v] = -3.0e38f; lrun[v] = 0.0f; }

    unsigned short* my = &plds[wave][0];

    for (int kt = 0; kt < q0 + 16; kt += 32) {
        // ---- scores: two 16x16 tiles over 32 keys, K-dim d=64 ----
        v8f s0 = {}, s1 = {};
        {
            const unsigned short* kr0 = kbase + (size_t)(kt + lm) * HD;
            const unsigned short* kr1 = kbase + (size_t)(kt + 16 + lm) * HD;
            v16bf b00 = load_frag(kr0 +      hi*8, kr0 + 16 + hi*8);
            v16bf b01 = load_frag(kr0 + 32 + hi*8, kr0 + 48 + hi*8);
            v16bf b10 = load_frag(kr1 +      hi*8, kr1 + 16 + hi*8);
            v16bf b11 = load_frag(kr1 + 32 + hi*8, kr1 + 48 + hi*8);
            s0 = wmma_bf16(qf0, b00, s0);
            s0 = wmma_bf16(qf1, b01, s0);
            s1 = wmma_bf16(qf0, b10, s1);
            s1 = wmma_bf16(qf1, b11, s1);
        }
        // ---- scale + causal mask + online softmax ----
#pragma unroll
        for (int v = 0; v < 8; ++v) {
            const int row  = q0 + v + hi*8;
            const int col0 = kt + lm;
            const int col1 = kt + 16 + lm;
            float e0 = s0[v] * 0.125f;
            float e1 = s1[v] * 0.125f;
            if (col0 > row) e0 = -3.0e38f;
            if (col1 > row) e1 = -3.0e38f;

            float mx = fmaxf(e0, e1);
#pragma unroll
            for (int off = 1; off < 16; off <<= 1)
                mx = fmaxf(mx, __shfl_xor(mx, off, 32));

            float mnew  = fmaxf(mrun[v], mx);
            float alpha = __expf(mrun[v] - mnew);
            float p0    = __expf(e0 - mnew);
            float p1    = __expf(e1 - mnew);
            float rs    = p0 + p1;
#pragma unroll
            for (int off = 1; off < 16; off <<= 1)
                rs += __shfl_xor(rs, off, 32);

            lrun[v] = lrun[v] * alpha + rs;
            mrun[v] = mnew;
#pragma unroll
            for (int jj = 0; jj < 4; ++jj) yacc[jj][v] *= alpha;

            // C-layout -> row-major 16x32 bf16 tile in LDS
            const int m = v + hi*8;
            my[m*32 + lm]      = f2bf(p0);
            my[m*32 + 16 + lm] = f2bf(p1);
        }
        asm volatile("s_wait_dscnt 0x0" ::: "memory");

        // ---- reload P as an A fragment (16x32, K-dim = keys) ----
        const unsigned short* prow = my + lm * 32;
        v16bf pf = load_frag(prow + hi*8, prow + 16 + hi*8);

        // ---- y += P * V ----
#pragma unroll
        for (int jj = 0; jj < 4; ++jj) {
            const unsigned short* vrow =
                vbase + (size_t)(jj*16 + lm) * Ss + kt;
            v16bf vf = load_frag(vrow + hi*8, vrow + 16 + hi*8);
            yacc[jj] = wmma_bf16(pf, vf, yacc[jj]);
        }
    }

    // ---- normalize and store Y (bf16, [B*S, H*64]) ----
#pragma unroll
    for (int jj = 0; jj < 4; ++jj)
#pragma unroll
        for (int v = 0; v < 8; ++v) {
            const int   row = q0 + v + hi*8;
            const float val = yacc[jj][v] / lrun[v];
            Y[((size_t)(b*Ss + row)) * DIM + h*HD + jj*16 + lm] = f2bf(val);
        }
}

// ---------------------------------------------------------------------------
// Host side
// ---------------------------------------------------------------------------
extern "C" void kernel_launch(void* const* d_in, const int* in_sizes, int n_in,
                              void* d_out, int out_size, void* d_ws, size_t ws_size,
                              hipStream_t stream) {
    (void)in_sizes; (void)n_in; (void)out_size; (void)ws_size;

    const float* x     = (const float*)d_in[0];   // [B,S,DIM]
    const float* freqs = (const float*)d_in[1];   // [S,32,2]
    //                         d_in[2] = causal mask (implicit, unused)
    const float* wqkv  = (const float*)d_in[3];   // [3*DIM, DIM]
    const float* wo    = (const float*)d_in[4];   // [DIM, DIM]
    float*       out   = (float*)d_out;           // [B,S,DIM]

    char* ws = (char*)d_ws;
    size_t off = 0;
    unsigned short* xbf    = (unsigned short*)(ws + off); off += (size_t)BS*DIM*2;      // 8 MB
    unsigned short* wqkvbf = (unsigned short*)(ws + off); off += (size_t)QKVN*DIM*2;    // 6 MB
    unsigned short* wobf   = (unsigned short*)(ws + off); off += (size_t)DIM*DIM*2;     // 2 MB
    float*          qkvf   = (float*)(ws + off);          off += (size_t)BS*QKVN*4;     // 48 MB
    unsigned short* qbf    = (unsigned short*)(ws + off); off += (size_t)BS*DIM*2;      // 8 MB
    unsigned short* kbf    = (unsigned short*)(ws + off); off += (size_t)BS*DIM*2;      // 8 MB
    unsigned short* vtbf   = (unsigned short*)(ws + off); off += (size_t)BS*DIM*2;      // 8 MB
    unsigned short* ybf    = (unsigned short*)(ws + off); off += (size_t)BS*DIM*2;      // 8 MB

    // 1) fp32 -> bf16
    k_to_bf16<<<(BS*DIM + 255)/256, 256, 0, stream>>>(x, xbf, BS*DIM);
    k_to_bf16<<<(QKVN*DIM + 255)/256, 256, 0, stream>>>(wqkv, wqkvbf, QKVN*DIM);
    k_to_bf16<<<(DIM*DIM + 255)/256, 256, 0, stream>>>(wo, wobf, DIM*DIM);

    // 2) QKV projection: [4096,1024] x [3072,1024]^T -> [4096,3072] f32
    k_gemm_bf16<<<dim3(QKVN/128, BS/64), 128, 0, stream>>>(
        xbf, wqkvbf, qkvf, BS, QKVN, DIM);

    // 3) RoPE + relayout
    k_rope<<<(BS*NH*32 + 255)/256, 256, 0, stream>>>(qkvf, freqs, qbf, kbf, vtbf);

    // 4) flash attention -> Y bf16 [4096, 1024]
    k_attn<<<512, 256, 0, stream>>>(qbf, kbf, vtbf, ybf);

    // 5) output projection: [4096,1024] x [1024,1024]^T -> out f32
    k_gemm_bf16<<<dim3(DIM/128, BS/64), 128, 0, stream>>>(
        ybf, wobf, out, BS, DIM, DIM);
}
